// TinyCondEpsNet_20813411516997
// MI455X (gfx1250) — matrix operand order repaired and checked by reading, not compiled, not run on hardware
//
#include <hip/hip_runtime.h>
#include <hip/hip_bf16.h>

typedef __attribute__((ext_vector_type(16))) _Float16 v16h;
typedef __attribute__((ext_vector_type(8)))  _Float16 v8h;
typedef __attribute__((ext_vector_type(4)))  _Float16 v4h;
typedef __attribute__((ext_vector_type(8)))  float    v8f;

#define BATCH      524288
#define XDIM       128
#define KDIM       161     // 1 + 128 + 32
#define KPAD       192     // 6 k-tiles of 32
#define HIDDEN     64
#define LDS_STRIDE 200     // f16 elems per row (400 B, 16B-aligned, bank-spread)
#define ROWS_PER_BLOCK 128
#define NKT 6
#define NNT 4

// Column permutation: new col k -> original col. x at 0..127, y_t at 128,
// te at 129..160 (unchanged), pad 161..191. Mirrored in weight reswizzle.
__device__ __forceinline__ int perm_col(int k) {
    if (k < 128) return k + 1;   // x
    if (k == 128) return 0;      // y_t
    if (k < 161) return k;       // te
    return -1;                   // zero pad
}

// Reswizzle fc1_weight [64,161] f32 -> f16 B-fragments in CDNA5 B layout.
// Word index i = ((ktile*4 + ntile)*32 + lane)*8 + v
// B layout (16x16x32 f16): VGPR v, lane l holds N = l%16, K = 16*(v>>2) + 2*(v&3) + 8*(l/16).
__global__ void prep_w1_kernel(const float* __restrict__ w1,
                               unsigned int* __restrict__ wsB) {
    int i = blockIdx.x * 256 + threadIdx.x;
    if (i >= NKT * NNT * 32 * 8) return;
    int v     = i & 7;
    int lane  = (i >> 3) & 31;
    int ntile = (i >> 8) & 3;
    int ktile = i >> 10;
    int n  = ntile * 16 + (lane & 15);
    int k0 = ktile * 32 + 16 * (v >> 2) + 2 * (v & 3) + 8 * (lane >> 4);
    int p0 = perm_col(k0);
    int p1 = perm_col(k0 + 1);
    float f0 = (p0 >= 0) ? w1[n * KDIM + p0] : 0.0f;
    float f1 = (p1 >= 0) ? w1[n * KDIM + p1] : 0.0f;
    _Float16 h0 = (_Float16)f0;
    _Float16 h1 = (_Float16)f1;
    unsigned short u0 = __builtin_bit_cast(unsigned short, h0);
    unsigned short u1 = __builtin_bit_cast(unsigned short, h1);
    wsB[i] = (unsigned int)u0 | ((unsigned int)u1 << 16);
}

__global__ __launch_bounds__(256)
void TinyCondEpsNet_mlp_kernel(const float* __restrict__ y_t,
                               const int*   __restrict__ t,
                               const float* __restrict__ x,
                               const float* __restrict__ b1,
                               const float* __restrict__ w2,
                               const float* __restrict__ b2,
                               const unsigned int* __restrict__ wsB,
                               float* __restrict__ out) {
    __shared__ _Float16 lds[ROWS_PER_BLOCK * LDS_STRIDE];

    const int tid = threadIdx.x;
    const int rowBase = blockIdx.x * ROWS_PER_BLOCK;

    // ---- Stage x (cols 0..127): coalesced float4 loads, 8B-aligned LDS stores
    const float4* x4 = (const float4*)x;
    #pragma unroll 4
    for (int i = tid; i < ROWS_PER_BLOCK * 32; i += 256) {
        int row = i >> 5;
        int c4  = i & 31;
        float4 vv = x4[(size_t)(rowBase + row) * 32 + c4];
        v4h h;
        h[0] = (_Float16)vv.x; h[1] = (_Float16)vv.y;
        h[2] = (_Float16)vv.z; h[3] = (_Float16)vv.w;
        *(v4h*)(&lds[row * LDS_STRIDE + c4 * 4]) = h;   // byte addr row*400 + c4*8
    }

    // ---- Stage y_t (col 128), sinusoidal embedding (cols 129..160), zero pad (161..199)
    for (int i = tid; i < ROWS_PER_BLOCK * 72; i += 256) {
        int row = i / 72;
        int c   = i - row * 72;          // 0..71 -> col 128..199
        float val;
        if (c == 0) {
            val = y_t[rowBase + row];
        } else if (c <= 32) {
            int j  = c - 1;              // te index 0..31
            int jj = j & 15;
            // freq = 10000^(-jj/16) = exp2(-jj * log2(10000)/16)
            float freq = exp2f(-(float)jj * 0.8304820237218407f);
            float arg  = (float)t[rowBase + row] * freq;
            val = (j < 16) ? sinf(arg) : cosf(arg);
        } else {
            val = 0.0f;
        }
        lds[row * LDS_STRIDE + 128 + c] = (_Float16)val;
    }

    __syncthreads();

    // ---- Per-wave 16x192 @ 192x64 matmul via 24x v_wmma_f32_16x16x32_f16
    const int wv   = tid >> 5;
    const int lane = tid & 31;
    const int m    = lane & 15;     // N index in C layout, M index in A layout
    const int hi   = lane >> 4;

    const _Float16* aRow = &lds[(wv * 16 + m) * LDS_STRIDE];

    v8f acc[NNT];
    #pragma unroll
    for (int nt = 0; nt < NNT; ++nt) acc[nt] = (v8f){};

    #pragma unroll
    for (int kt = 0; kt < NKT; ++kt) {
        // A fragment, 16-bit A layout: lane(m,hi), VGPR v -> K = 16*(v>>2) + 2*(v&3) + 8*hi
        // => two contiguous 16B chunks at f16 offsets kt*32 + 8*hi and +16.
        v8h a0 = *(const v8h*)(aRow + kt * 32 + 8 * hi);
        v8h a1 = *(const v8h*)(aRow + kt * 32 + 8 * hi + 16);
        v16h a = __builtin_shufflevector(a0, a1,
                                         0, 1, 2, 3, 4, 5, 6, 7,
                                         8, 9, 10, 11, 12, 13, 14, 15);
        #pragma unroll
        for (int nt = 0; nt < NNT; ++nt) {
            const v16h b = *(const v16h*)(wsB + ((size_t)((kt * NNT + nt) * 32 + lane)) * 8);
            acc[nt] = __builtin_amdgcn_wmma_f32_16x16x32_f16(
                false, a, false, b, (short)0, acc[nt], false, false);
        }
    }

    // ---- Epilogue: bias + ReLU + fc2 dot, in registers.
    // C layout: VGPR r, lane -> M = r + 8*hi, N = m.
    float partial[8];
    #pragma unroll
    for (int r = 0; r < 8; ++r) partial[r] = 0.0f;

    #pragma unroll
    for (int nt = 0; nt < NNT; ++nt) {
        float bn = b1[nt * 16 + m];
        float wn = w2[nt * 16 + m];
        #pragma unroll
        for (int r = 0; r < 8; ++r) {
            float h = acc[nt][r] + bn;
            h = (h > 0.0f) ? h : 0.0f;
            partial[r] += h * wn;
        }
    }

    // Reduce across the 16 lanes sharing each row (wave32 xor shuffles stay in-half)
    #pragma unroll
    for (int r = 0; r < 8; ++r) {
        float s = partial[r];
        s += __shfl_xor(s, 1, 32);
        s += __shfl_xor(s, 2, 32);
        s += __shfl_xor(s, 4, 32);
        s += __shfl_xor(s, 8, 32);
        partial[r] = s;
    }

    if (m == 0) {
        float bb = b2[0];
        #pragma unroll
        for (int r = 0; r < 8; ++r) {
            out[rowBase + wv * 16 + hi * 8 + r] = partial[r] + bb;
        }
    }
}

extern "C" void kernel_launch(void* const* d_in, const int* in_sizes, int n_in,
                              void* d_out, int out_size, void* d_ws, size_t ws_size,
                              hipStream_t stream) {
    const float* y_t = (const float*)d_in[0];
    const int*   t   = (const int*)  d_in[1];
    const float* x   = (const float*)d_in[2];
    const float* w1  = (const float*)d_in[3];
    const float* b1  = (const float*)d_in[4];
    const float* w2  = (const float*)d_in[5];
    const float* b2  = (const float*)d_in[6];
    unsigned int* wsB = (unsigned int*)d_ws;   // 6*4*32*8 dwords = 24 KB
    float* out = (float*)d_out;

    prep_w1_kernel<<<(NKT * NNT * 32 * 8 + 255) / 256, 256, 0, stream>>>(w1, wsB);
    TinyCondEpsNet_mlp_kernel<<<BATCH / ROWS_PER_BLOCK, 256, 0, stream>>>(
        y_t, t, x, b1, w2, b2, wsB, out);
}
